// pLSTM_36833639530918
// MI455X (gfx1250) — matrix operand order repaired and checked by reading, not compiled, not run on hardware
//
#include <hip/hip_runtime.h>
#include <hip/hip_bf16.h>
#include <math.h>

// Problem dims
#define BB   64
#define SS   512
#define FF   512
#define HH   1024
#define N3   3072          // 3*H  (r | c | o gates)
#define KDIM 1536          // F + H (fused [x_t | h_{t-1}] GEMM)
#define EPS  1e-3f

typedef __attribute__((ext_vector_type(16))) __bf16 v16bf;
typedef __attribute__((ext_vector_type(8)))  float  v8f;

union Frag {
    v16bf v;
    uint4 q[2];
};

// ---------------------------------------------------------------------------
// Prep: Wcat^T in bf16, column-major [N=3072][K=1536] so B-fragment loads are
// contiguous 16B vectors. Rows 0..511 of K come from U_*, 512..1535 from W_*.
// ---------------------------------------------------------------------------
__global__ void __launch_bounds__(256)
prep_w(const float* __restrict__ Ur, const float* __restrict__ Wr,
       const float* __restrict__ Uc, const float* __restrict__ Wc,
       const float* __restrict__ Uo, const float* __restrict__ Wo,
       __bf16* __restrict__ WT) {
    int idx = blockIdx.x * blockDim.x + threadIdx.x;   // N3*KDIM threads
    if (idx >= N3 * KDIM) return;
    int n = idx / KDIM;
    int k = idx - n * KDIM;
    int sel = n >> 10;          // 0=r, 1=c, 2=o
    int h   = n & (HH - 1);
    const float* Um = (sel == 0) ? Ur : (sel == 1) ? Uc : Uo;
    const float* Wm = (sel == 0) ? Wr : (sel == 1) ? Wc : Wo;
    float val = (k < FF) ? Um[(size_t)k * HH + h]
                         : Wm[(size_t)(k - FF) * HH + h];
    WT[idx] = (__bf16)val;
}

// Convert x to bf16 once (halves loop-body A-operand traffic).
__global__ void __launch_bounds__(256)
prep_x(const float* __restrict__ x, __bf16* __restrict__ xb, int n) {
    int i = blockIdx.x * blockDim.x + threadIdx.x;
    if (i < n) xb[i] = (__bf16)x[i];
}

// Zero-init recurrent state (workspace is poisoned by the harness).
__global__ void __launch_bounds__(256)
init_state(__bf16* __restrict__ hstate, float* __restrict__ cstate,
           float* __restrict__ kstate) {
    int i = blockIdx.x * blockDim.x + threadIdx.x;     // B*H threads
    hstate[i] = (__bf16)0.0f;
    cstate[i] = 0.0f;
    kstate[i] = 0.0f;
}

// ---------------------------------------------------------------------------
// Per-step fused GEMM:  G[64,3072] = [x_t | h_{t-1}] (64x1536) @ WcatT^T
// One 16x16 tile per wave; 4 M-tiles x 192 N-tiles = 768 waves.
// v_wmma_f32_16x16x32_bf16, K-loop of 48.
// ---------------------------------------------------------------------------
__device__ __forceinline__ v8f
wmma_step(const __bf16* __restrict__ ap, const __bf16* __restrict__ bp,
          int half, v8f acc) {
    Frag a, b;
    // A 16x32 bf16 layout: lanes0-15 row M, K groups {half*8.., 16+half*8..}
    a.q[0] = *(const uint4*)(ap + half * 8);
    a.q[1] = *(const uint4*)(ap + 16 + half * 8);
    // B 32x16 bf16 layout: lane = N column, K groups {half*8.., 16+half*8..}
    b.q[0] = *(const uint4*)(bp + half * 8);
    b.q[1] = *(const uint4*)(bp + 16 + half * 8);
    return __builtin_amdgcn_wmma_f32_16x16x32_bf16(
        /*neg_a=*/false, a.v, /*neg_b=*/false, b.v,
        /*c_mod=*/(short)0, acc, /*reuse_a=*/false, /*reuse_b=*/false);
}

__global__ void __launch_bounds__(128)
gemm_step(const __bf16* __restrict__ xb, const __bf16* __restrict__ hstate,
          const __bf16* __restrict__ WT, float* __restrict__ G, int step) {
    int wave   = blockIdx.x * (blockDim.x >> 5) + (threadIdx.x >> 5);
    int lane   = threadIdx.x & 31;
    int half   = lane >> 4;
    int lm     = lane & 15;
    int tile_m = wave & 3;        // 0..3   (M = 64)
    int tile_n = wave >> 2;       // 0..191 (N = 3072)

    int mrow = tile_m * 16 + lm;
    int ncol = tile_n * 16 + lm;

    const __bf16* arow_x = xb + ((size_t)mrow * SS + step) * FF;
    const __bf16* arow_h = hstate + (size_t)mrow * HH;
    const __bf16* bcol   = WT + (size_t)ncol * KDIM;

    v8f acc = {};
#pragma unroll 8
    for (int kk = 0; kk < FF / 32; ++kk) {            // x_t part (K 0..511)
        int kb = kk * 32;
        acc = wmma_step(arow_x + kb, bcol + kb, half, acc);
    }
#pragma unroll 8
    for (int kk = 0; kk < HH / 32; ++kk) {            // h_{t-1} part (K 512..1535)
        int kb = kk * 32;
        acc = wmma_step(arow_h + kb, bcol + FF + kb, half, acc);
    }

    // C/D layout: VGPR r -> row (r + half*8); lane 0-15 -> column
    int col = tile_n * 16 + lm;
#pragma unroll
    for (int r = 0; r < 8; ++r)
        G[(size_t)(tile_m * 16 + half * 8 + r) * N3 + col] = acc[r];
}

// ---------------------------------------------------------------------------
// Per-step gate math (phased-LSTM update), writes h_t to output + bf16 state.
// ---------------------------------------------------------------------------
__global__ void __launch_bounds__(256)
epilogue(const float* __restrict__ G, const float* __restrict__ br,
         const float* __restrict__ bc, const float* __restrict__ bo,
         const float* __restrict__ timeseq, __bf16* __restrict__ hstate,
         float* __restrict__ cstate, float* __restrict__ kstate,
         float* __restrict__ out, int step) {
    int idx = blockIdx.x * blockDim.x + threadIdx.x;   // B*H threads
    int b = idx >> 10;
    int h = idx & (HH - 1);

    float gr = G[(size_t)b * N3 + h]          + br[h];
    float gc = G[(size_t)b * N3 + HH + h]     + bc[h];
    float go = G[(size_t)b * N3 + 2 * HH + h] + bo[h];
    float t  = timeseq[(size_t)b * SS + step];

    float r  = 1.0f / (1.0f + __expf(-gr));
    float kn = r * t + (1.0f - r) * kstate[idx];
    float ct = tanhf(gc);
    float d  = t - kn;
    // f = ((d+1)/(d+eps))^(-1/2) = sqrt((d+eps)/(d+1))
    float f  = __fsqrt_rn((d + EPS) / (d + 1.0f));
    float cn = f * cstate[idx] + (1.0f - f) * ct;
    float o  = 1.0f / (1.0f + __expf(-go));
    float hn = o * tanhf(cn);

    kstate[idx] = kn;
    cstate[idx] = cn;
    hstate[idx] = (__bf16)hn;
    out[((size_t)b * SS + step) * HH + h] = hn;

    if (step == SS - 1) {                              // final (h,c,k) states
        size_t base = (size_t)BB * SS * HH;
        out[base + idx]               = hn;
        out[base + BB * HH + idx]     = cn;
        out[base + 2 * BB * HH + idx] = kn;
    }
}

// ---------------------------------------------------------------------------
extern "C" void kernel_launch(void* const* d_in, const int* in_sizes, int n_in,
                              void* d_out, int out_size, void* d_ws, size_t ws_size,
                              hipStream_t stream) {
    const float* x  = (const float*)d_in[0];
    const float* ts = (const float*)d_in[1];
    const float* Ur = (const float*)d_in[2];
    const float* Wr = (const float*)d_in[3];
    const float* br = (const float*)d_in[4];
    const float* Uc = (const float*)d_in[5];
    const float* Wc = (const float*)d_in[6];
    const float* bc = (const float*)d_in[7];
    const float* Uo = (const float*)d_in[8];
    const float* Wo = (const float*)d_in[9];
    const float* bo = (const float*)d_in[10];

    char* ws = (char*)d_ws;
    size_t off = 0;
    __bf16* WT     = (__bf16*)(ws + off); off += (size_t)N3 * KDIM * 2;     // 9.4 MB
    __bf16* xb     = (__bf16*)(ws + off); off += (size_t)BB * SS * FF * 2;  // 32 MB
    __bf16* hstate = (__bf16*)(ws + off); off += (size_t)BB * HH * 2;
    float*  cstate = (float*) (ws + off); off += (size_t)BB * HH * 4;
    float*  kstate = (float*) (ws + off); off += (size_t)BB * HH * 4;
    float*  G      = (float*) (ws + off);                                   // 0.75 MB
    float*  out    = (float*)d_out;

    prep_w<<<(N3 * KDIM + 255) / 256, 256, 0, stream>>>(Ur, Wr, Uc, Wc, Uo, Wo, WT);
    prep_x<<<(BB * SS * FF + 255) / 256, 256, 0, stream>>>(x, xb, BB * SS * FF);
    init_state<<<(BB * HH) / 256, 256, 0, stream>>>(hstate, cstate, kstate);

    for (int step = 0; step < SS; ++step) {
        gemm_step<<<192, 128, 0, stream>>>(xb, hstate, WT, G, step);
        epilogue<<<(BB * HH) / 256, 256, 0, stream>>>(G, br, bc, bo, ts,
                                                      hstate, cstate, kstate,
                                                      out, step);
    }
}
